// BasicNCAModel_20203526160773
// MI455X (gfx1250) — matrix operand order repaired and checked by reading, not compiled, not run on hardware
//
#include <hip/hip_runtime.h>
#include <hip/hip_bf16.h>

// ---------------------------------------------------------------------------
// NCA fused step on gfx1250: perceive (depthwise convs) + 2 GEMMs via
// v_wmma_f32_16x16x32_bf16, f32 accumulate, masked residual update.
// ---------------------------------------------------------------------------

#define Cc    24      // channels
#define NIC   4       // immutable image channels
#define HID   128     // hidden
#define K1    72      // 3*C
#define K1P   96      // K1 padded to 3*32
#define PHS   136     // row stride (bf16) of shared p/h buffer: 272B = 16B*17, bank-clean
#define TH    8       // tile height (pixels)
#define TW    16      // tile width  (pixels)

typedef __attribute__((ext_vector_type(16))) __bf16 v16bf;
typedef __attribute__((ext_vector_type(8)))  float  v8f;

struct U4x2 { uint4 lo, hi; };

__device__ __forceinline__ unsigned short f2bf(float f) {
    unsigned int u = __builtin_bit_cast(unsigned int, f);
    u += 0x7FFFu + ((u >> 16) & 1u);          // round-to-nearest-even
    return (unsigned short)(u >> 16);
}

// A-fragment (16x32 bf16, M x K) from LDS row-major [m][k] with row stride PHS.
// Per ISA table: lanes 0-15 hold K = {kb..kb+7, kb+16..kb+23}, lanes 16-31 hold
// K = {kb+8..kb+15, kb+24..kb+31}; row m = lane%16. Two aligned b128 loads.
__device__ __forceinline__ v16bf load_afrag(const unsigned short* rowbase, int lane, int kb) {
    const int m  = lane & 15;
    const int h8 = (lane >> 4) << 3;
    const unsigned short* r = rowbase + m * PHS;
    U4x2 t;
    t.lo = *(const uint4*)(r + kb + h8);
    t.hi = *(const uint4*)(r + kb + 16 + h8);
    return __builtin_bit_cast(v16bf, t);
}

// B-fragment (32x16 bf16, K x N) from a row-major [N][ks] bf16 weight array
// (so B[k][n] = W[n][k]). Lanes 0-15: K = {kb..kb+7, kb+16..kb+23}; lanes 16-31: K = {kb+8..kb+15, kb+24..kb+31};
// column n = n0 + lane%16. Two aligned b128 loads per lane.
__device__ __forceinline__ v16bf load_bfrag(const unsigned short* W, int ks, int lane, int n0, int kb) {
    const int n = n0 + (lane & 15);
    const int k = kb + ((lane >> 4) << 3);
    const unsigned short* p = W + n * ks + k;
    U4x2 t;
    t.lo = *(const uint4*)(p);
    t.hi = *(const uint4*)(p + 16);
    return __builtin_bit_cast(v16bf, t);
}

// --------------------------- weight prep ----------------------------------
__global__ void nca_prep(const float* __restrict__ W1, const float* __restrict__ W2,
                         unsigned short* __restrict__ W1p, unsigned short* __restrict__ W2p) {
    int i = blockIdx.x * 256 + threadIdx.x;
    if (i < HID * K1P) {                       // W1p: [128][96] bf16, K padded w/ 0
        int n = i / K1P, k = i % K1P;
        W1p[i] = (k < K1) ? f2bf(W1[n * K1 + k]) : (unsigned short)0;
    }
    if (i < 32 * HID) {                        // W2p: [32][128] bf16, rows 24..31 = 0
        int c = i / HID, k = i % HID;
        W2p[i] = (c < Cc) ? f2bf(W2[c * HID + k]) : (unsigned short)0;
    }
}

// --------------------------- fused NCA step -------------------------------
__global__ __launch_bounds__(256, 1)
void nca_step(const float* __restrict__ src, float* __restrict__ dst,
              const float* __restrict__ w1g, const float* __restrict__ w2g,
              const unsigned short* __restrict__ W1p, const unsigned short* __restrict__ W2p,
              const float* __restrict__ b1, const float* __restrict__ rand_u, int step) {
    __shared__ __align__(16) float          xh[10 * 18 * Cc];   // halo tile  (17280B)
    __shared__ __align__(16) float          wcv[2 * 216];       // w1 | w2    (1728B)
    __shared__ __align__(16) float          fire[TH * TW];      // per-pixel fire mask
    __shared__ __align__(16) unsigned short ph[TH * TW * PHS];  // p (K=96) then h (K=128), bf16

    const int tid  = threadIdx.x;
    const int b    = blockIdx.x >> 7;          // image
    const int tidx = blockIdx.x & 127;
    const int r0   = (tidx >> 3) * TH;         // tile origin row
    const int c0   = (tidx & 7)  * TW;         // tile origin col

    // ---- stage halo [10][18][24] f32 with b128 loads (circular wrap) ----
    for (int i = tid; i < 10 * 18 * (Cc / 4); i += 256) {
        int pix = i / 6, cq = i % 6;
        int row = pix / 18, col = pix % 18;
        int gy = (r0 + row - 1) & 127;
        int gx = (c0 + col - 1) & 127;
        const float4 v = *(const float4*)(src + ((size_t)((b * 128 + gy) * 128 + gx)) * Cc + cq * 4);
        *(float4*)(xh + (row * 18 + col) * Cc + cq * 4) = v;
    }
    for (int i = tid; i < 432; i += 256) wcv[i] = (i < 216) ? w1g[i] : w2g[i - 216];
    if (tid < TH * TW) {
        int py = tid >> 4, px = tid & 15;
        float u = rand_u[(((size_t)step * 8 + b) * 128 + (r0 + py)) * 128 + (c0 + px)];
        fire[tid] = (u < 0.5f) ? 1.0f : 0.0f;
    }
    __syncthreads();

    // ---- perceive: identity + 2 depthwise 3x3 convs -> bf16 p[m][0..95] ----
    {
        const int m  = tid >> 1;
        const int hh = tid & 1;
        const int py = m >> 4, px = m & 15;
        unsigned short* prow = ph + m * PHS;
        #pragma unroll
        for (int j = 0; j < 12; ++j) {
            const int ch = hh * 12 + j;
            float a1 = 0.f, a2 = 0.f;
            #pragma unroll
            for (int dy = 0; dy < 3; ++dy)
                #pragma unroll
                for (int dx = 0; dx < 3; ++dx) {
                    float xv = xh[((py + dy) * 18 + (px + dx)) * Cc + ch];
                    a1 = fmaf(wcv[(dy * 3 + dx) * Cc + ch],       xv, a1);
                    a2 = fmaf(wcv[216 + (dy * 3 + dx) * Cc + ch], xv, a2);
                }
            prow[ch]          = f2bf(xh[((py + 1) * 18 + (px + 1)) * Cc + ch]);
            prow[Cc + ch]     = f2bf(a1);
            prow[2 * Cc + ch] = f2bf(a2);
            prow[K1 + hh * 12 + j] = 0;        // zero K padding 72..95
        }
    }
    __syncthreads();

    // ---- wave-local GEMMs: each wave owns a 16-row M-tile ----
    const int wv   = tid >> 5;
    const int lane = tid & 31;
    const int m0   = wv << 4;
    const unsigned short* myrows = ph + m0 * PHS;

    // layer1: [16 x 96] x [96 x 128] -> h[16 x 128], f32 accum
    v8f acc[8];
    #pragma unroll
    for (int nt = 0; nt < 8; ++nt) acc[nt] = (v8f){};
    #pragma unroll
    for (int kk = 0; kk < 3; ++kk) {
        const v16bf A = load_afrag(myrows, lane, kk * 32);
        #pragma unroll
        for (int nt = 0; nt < 8; ++nt) {
            const v16bf Bv = load_bfrag(W1p, K1P, lane, nt * 16, kk * 32);
            acc[nt] = __builtin_amdgcn_wmma_f32_16x16x32_bf16(
                false, A, false, Bv, (short)0, acc[nt], false, false);
        }
    }

    // bias + relu, h -> bf16 back into same LDS rows (wave-local, DS in-order)
    {
        const int mb = m0 + ((lane >> 4) << 3);
        #pragma unroll
        for (int nt = 0; nt < 8; ++nt) {
            const int n = nt * 16 + (lane & 15);
            const float bias = b1[n];
            #pragma unroll
            for (int r = 0; r < 8; ++r) {
                float hv = acc[nt][r] + bias;
                hv = hv > 0.f ? hv : 0.f;
                ph[(mb + r) * PHS + n] = f2bf(hv);
            }
        }
    }

    // layer2: [16 x 128] x [128 x 32] -> dx[16 x 32] (cols >= 24 discarded)
    v8f acc2[2];
    acc2[0] = (v8f){}; acc2[1] = (v8f){};
    #pragma unroll
    for (int kk = 0; kk < 4; ++kk) {
        const v16bf A = load_afrag(myrows, lane, kk * 32);
        #pragma unroll
        for (int nt = 0; nt < 2; ++nt) {
            const v16bf Bv = load_bfrag(W2p, HID, lane, nt * 16, kk * 32);
            acc2[nt] = __builtin_amdgcn_wmma_f32_16x16x32_bf16(
                false, A, false, Bv, (short)0, acc2[nt], false, false);
        }
    }

    // ---- masked residual update ----
    {
        const int cl = lane & 15;
        const int mb = m0 + ((lane >> 4) << 3);
        #pragma unroll
        for (int nt = 0; nt < 2; ++nt) {
            const int c = nt * 16 + cl;
            if (c < Cc) {
                const float chm = (c >= NIC) ? 1.0f : 0.0f;
                #pragma unroll
                for (int r = 0; r < 8; ++r) {
                    const int m  = mb + r;
                    const int py = m >> 4, px = m & 15;
                    const float xo = xh[((py + 1) * 18 + (px + 1)) * Cc + c];
                    const float d  = acc2[nt][r] * fire[m] * chm;
                    dst[((size_t)((b * 128 + (r0 + py)) * 128 + (c0 + px))) * Cc + c] = xo + d;
                }
            }
        }
    }
}

// ---------------------------------------------------------------------------
extern "C" void kernel_launch(void* const* d_in, const int* in_sizes, int n_in,
                              void* d_out, int out_size, void* d_ws, size_t ws_size,
                              hipStream_t stream) {
    const float* x   = (const float*)d_in[0];
    const float* w1  = (const float*)d_in[1];
    const float* w2  = (const float*)d_in[2];
    const float* W1  = (const float*)d_in[3];
    const float* b1  = (const float*)d_in[4];
    const float* W2  = (const float*)d_in[5];
    const float* ru  = (const float*)d_in[6];
    // steps is fixed at 8 per the reference setup (cannot read device scalar
    // on host inside graph capture).
    const int steps = 8;

    unsigned short* W1p = (unsigned short*)d_ws;                      // 128*96 bf16
    unsigned short* W2p = W1p + HID * K1P;                            // 32*128 bf16
    size_t woff = ((size_t)(HID * K1P + 32 * HID) * 2 + 255) & ~(size_t)255;
    float* buf0 = (float*)((char*)d_ws + woff);                       // ping buffer
    float* out  = (float*)d_out;

    nca_prep<<<(HID * K1P + 255) / 256, 256, 0, stream>>>(W1, W2, W1p, W2p);

    const float* s = x;
    for (int i = 0; i < steps; ++i) {
        float* dptr = (i == steps - 1) ? out : ((i & 1) ? out : buf0);
        nca_step<<<8 * 128, 256, 0, stream>>>(s, dptr, w1, w2, W1p, W2p, b1, ru, i);
        s = dptr;
    }
}
